// SMPLX_27410481283809
// MI455X (gfx1250) — matrix operand order repaired and checked by reading, not compiled, not run on hardware
//
#include <hip/hip_runtime.h>
#include <math.h>

// ---------------------------------------------------------------------------
// SMPL-X forward (LBS) for MI455X / gfx1250, wave32.
//  - bf16 WMMA 16x16x32 (f32 accumulate) for both big GEMMs
//  - Tensor Data Mover (tensor_load_to_lds) double-buffered LDS staging of the
//    posedirs B-panel, tracked with TENSORcnt
// ---------------------------------------------------------------------------

#define VNUM 10475
#define JNUM 55
#define BNUM 512
#define N3   31425          // V*3
#define NP2  31488          // N3 padded to 128 (128*246)
#define KP   512            // 486 padded to 512 (8 chunks of 64)
#define VP   10480          // V padded to 16   (16*655)
#define VTILES 655
#define NGROUPS 246         // NP2 / 128
#define MGROUPS 8           // 32 m-tiles / 4 per wave

typedef __attribute__((ext_vector_type(8)))  float  v8f;
typedef __attribute__((ext_vector_type(8)))  __bf16 bf16x8;
typedef __attribute__((ext_vector_type(16))) __bf16 bf16x16;
typedef __attribute__((ext_vector_type(4)))  unsigned int v4u;
typedef __attribute__((ext_vector_type(8)))  int v8i;
typedef __attribute__((ext_vector_type(4)))  int v4i;

__device__ __forceinline__ bf16x16 load_frag(const __bf16* p0, const __bf16* p1) {
    union { bf16x16 v; bf16x8 h[2]; } u;
    u.h[0] = *reinterpret_cast<const bf16x8*>(p0);
    u.h[1] = *reinterpret_cast<const bf16x8*>(p1);
    return u.v;
}

// TDM: DMA a (128 rows x 64 K-elements) bf16 tile from the K-major posedirs
// panel into LDS. D# built per CDNA5 ISA 8.3-8.6. 2D tile, data_size=2B,
// row stride = KP elements. Issued by one wave, tracked with TENSORcnt.
// This toolchain exposes the 6-arg builtin (g0, g1, g2, g3, g-extra, cpol).
__device__ __forceinline__ void tdm_load_tile(unsigned long long gaddr, unsigned ldsaddr) {
    v4u g0 = { 1u,                               // count=1, is_restore=0
               ldsaddr,                          // lds_addr [63:32]
               (unsigned)(gaddr & 0xffffffffu),  // global_addr lo
               (unsigned)(gaddr >> 32) | (2u << 30) };  // addr hi | type=2
    v8i g1;
    g1[0] = 0x00010000;                 // wg_mask=0, data_size=1 (2 bytes)
    g1[1] = (int)(512u   << 16);        // tensor_dim0 = KP (lo16 @ bit48)
    g1[2] = (int)(31488u << 16);        // tensor_dim0 hi=0 | tensor_dim1 lo16 @ bit80
    g1[3] = (int)(64u    << 16);        // tensor_dim1 hi=0 | tile_dim0 = 64 @ bit112
    g1[4] = 128;                        // tile_dim1 = 128, tile_dim2 = 0
    g1[5] = 512;                        // tensor_dim0_stride = KP
    g1[6] = 0;
    g1[7] = 0;
    v4i z4 = { 0, 0, 0, 0 };
    v8i z8;
#pragma unroll
    for (int q = 0; q < 8; ++q) z8[q] = 0;
    __builtin_amdgcn_tensor_load_to_lds(g0, g1, z4, z4, z8, 0);
}

// ---- k_base: base_shaped[v*3+c] = v_template + sum_s shapedirs*shape[s] ----
__global__ void k_base(const float* __restrict__ vt, const float* __restrict__ sd,
                       const float* __restrict__ shp, float* __restrict__ base) {
    int i = blockIdx.x * 256 + threadIdx.x;
    if (i >= N3) return;
    float acc = vt[i];
    const float* s = sd + (size_t)i * 10;
#pragma unroll
    for (int e = 0; e < 10; ++e) acc += s[e] * shp[e];
    base[i] = acc;
}

// ---- k_yoff: y_offset = -min(v_template[:,1]) ----
__global__ void k_yoff(const float* __restrict__ vt, float* __restrict__ yoff) {
    __shared__ float s[256];
    float m = 3.4e38f;
    for (int v = threadIdx.x; v < VNUM; v += 256) m = fminf(m, vt[v * 3 + 1]);
    s[threadIdx.x] = m; __syncthreads();
    for (int st = 128; st > 0; st >>= 1) {
        if (threadIdx.x < st) s[threadIdx.x] = fminf(s[threadIdx.x], s[threadIdx.x + st]);
        __syncthreads();
    }
    if (threadIdx.x == 0) yoff[0] = -s[0];
}

// ---- k_jreg: block = (j,k); k<3 -> joints_base, else JE = JR @ exprdirs ----
__global__ void k_jreg(const float* __restrict__ JR, const float* __restrict__ base,
                       const float* __restrict__ ed, float* __restrict__ jbase,
                       float* __restrict__ JE) {
    int j = blockIdx.x / 33, k = blockIdx.x % 33;
    __shared__ float s[256];
    float acc = 0.f;
    for (int v = threadIdx.x; v < VNUM; v += 256) {
        float w = JR[(size_t)j * VNUM + v];
        float x = (k < 3) ? base[v * 3 + k] : ed[(size_t)v * 30 + (k - 3)];
        acc += w * x;
    }
    s[threadIdx.x] = acc; __syncthreads();
    for (int st = 128; st > 0; st >>= 1) {
        if (threadIdx.x < st) s[threadIdx.x] += s[threadIdx.x + st];
        __syncthreads();
    }
    if (threadIdx.x == 0) {
        if (k < 3) jbase[j * 3 + k] = s[0];
        else       JE[j * 30 + (k - 3)] = s[0];
    }
}

// ---- k_pose: rodrigues + per-batch joints ----
__global__ void k_pose(const float* __restrict__ pelvis, const float* __restrict__ body,
                       const float* __restrict__ head, const float* __restrict__ hand,
                       const float* __restrict__ hmean, const float* __restrict__ expr,
                       const float* __restrict__ jbase, const float* __restrict__ JE,
                       float* __restrict__ rotR, float* __restrict__ joints) {
    int i = blockIdx.x * 256 + threadIdx.x;
    if (i >= BNUM * JNUM) return;
    int b = i / JNUM, j = i % JNUM;
    float r0, r1, r2;
    if (j == 0)      { r0 = pelvis[b*3];              r1 = pelvis[b*3+1];            r2 = pelvis[b*3+2]; }
    else if (j <= 21){ int o = b*63 + (j-1)*3;        r0 = body[o]; r1 = body[o+1];  r2 = body[o+2]; }
    else if (j <= 24){ int o = b*9 + (j-22)*3;        r0 = head[o]; r1 = head[o+1];  r2 = head[o+2]; }
    else { int h = j - 25; int o = b*90 + h*3;
           r0 = hand[o]   + hmean[h*3];
           r1 = hand[o+1] + hmean[h*3+1];
           r2 = hand[o+2] + hmean[h*3+2]; }
    float a2  = r0*r0 + r1*r1 + r2*r2 + 1e-12f;
    float ang = sqrtf(a2);
    float inv = 1.f / ang;
    float kx = r0*inv, ky = r1*inv, kz = r2*inv;
    float s = sinf(ang), c = cosf(ang), t = 1.f - c;
    float R[9];
    R[0] = c + t*kx*kx;     R[1] = t*kx*ky - s*kz;  R[2] = t*kx*kz + s*ky;
    R[3] = t*kx*ky + s*kz;  R[4] = c + t*ky*ky;     R[5] = t*ky*kz - s*kx;
    R[6] = t*kx*kz - s*ky;  R[7] = t*ky*kz + s*kx;  R[8] = c + t*kz*kz;
    float* rp = rotR + (size_t)b * (JNUM*9) + j*9;
#pragma unroll
    for (int q = 0; q < 9; ++q) rp[q] = R[q];
    const float* ex = expr + b*10;
#pragma unroll
    for (int cc = 0; cc < 3; ++cc) {
        float jv = jbase[j*3 + cc];
        const float* je = JE + j*30 + cc*10;
#pragma unroll
        for (int e = 0; e < 10; ++e) jv += je[e] * ex[e];
        joints[(size_t)b * (JNUM*3) + j*3 + cc] = jv;
    }
}

// ---- k_fk: per-batch sequential kinematic chain (PARENTS = chain) ----
__global__ void k_fk(const float* __restrict__ rotR, const float* __restrict__ joints,
                     __bf16* __restrict__ At) {
    int b = blockIdx.x * 256 + threadIdx.x;
    if (b >= BNUM) return;
    const float* R  = rotR   + (size_t)b * (JNUM*9);
    const float* Jt = joints + (size_t)b * (JNUM*3);
    __bf16* at = At + (size_t)b * 1024;       // (16 components) x (64 joints)
    for (int q = 0; q < 1024; ++q) at[q] = (__bf16)0.f;
    float G[12];
#pragma unroll
    for (int r = 0; r < 3; ++r) {
        G[r*4+0] = R[r*3+0]; G[r*4+1] = R[r*3+1]; G[r*4+2] = R[r*3+2];
        G[r*4+3] = Jt[r];
    }
    for (int j = 0; j < JNUM; ++j) {
        if (j > 0) {
            const float* Rj = R + j*9;
            float t0 = Jt[j*3]   - Jt[j*3-3];
            float t1 = Jt[j*3+1] - Jt[j*3-2];
            float t2 = Jt[j*3+2] - Jt[j*3-1];
            float Nn[12];
#pragma unroll
            for (int r = 0; r < 3; ++r) {
                float g0 = G[r*4], g1 = G[r*4+1], g2 = G[r*4+2];
#pragma unroll
                for (int cc = 0; cc < 3; ++cc)
                    Nn[r*4+cc] = g0*Rj[cc] + g1*Rj[3+cc] + g2*Rj[6+cc];
                Nn[r*4+3] = g0*t0 + g1*t1 + g2*t2 + G[r*4+3];
            }
#pragma unroll
            for (int q = 0; q < 12; ++q) G[q] = Nn[q];
        }
        float jx = Jt[j*3], jy = Jt[j*3+1], jz = Jt[j*3+2];
#pragma unroll
        for (int r = 0; r < 3; ++r) {
            float tr = G[r*4+3] - (G[r*4]*jx + G[r*4+1]*jy + G[r*4+2]*jz);
            at[(r*4+0)*64 + j] = (__bf16)G[r*4+0];
            at[(r*4+1)*64 + j] = (__bf16)G[r*4+1];
            at[(r*4+2)*64 + j] = (__bf16)G[r*4+2];
            at[(r*4+3)*64 + j] = (__bf16)tr;
        }
    }
}

// ---- k_pdT: posedirs (486 x N3) f32 -> K-major bf16 (NP2 x KP), LDS tiled ----
__global__ void k_pdT(const float* __restrict__ pd, __bf16* __restrict__ pdt) {
    __shared__ float t[32][33];
    int nt = blockIdx.x * 32, kt = blockIdx.y * 32;
    int tx = threadIdx.x, ty = threadIdx.y;     // block (32, 8)
#pragma unroll
    for (int i = 0; i < 4; ++i) {
        int k = kt + ty + i*8, n = nt + tx;
        t[ty + i*8][tx] = (k < 486 && n < N3) ? pd[(size_t)k * N3 + n] : 0.f;
    }
    __syncthreads();
#pragma unroll
    for (int i = 0; i < 4; ++i) {
        int n = nt + ty + i*8, k = kt + tx;
        if (n < NP2) pdt[(size_t)n * KP + k] = (__bf16)t[tx][ty + i*8];
    }
}

// ---- k_apose: pose_feature (rot - I) -> bf16 (B x KP) zero-padded ----
__global__ void k_apose(const float* __restrict__ rotR, __bf16* __restrict__ Ap) {
    int i = blockIdx.x * 256 + threadIdx.x;
    if (i >= BNUM * KP) return;
    int b = i / KP, k = i % KP;
    float v = 0.f;
    if (k < 486) {
        int j = k / 9 + 1, q = k % 9;
        v = rotR[(size_t)b * (JNUM*9) + j*9 + q];
        if (q == 0 || q == 4 || q == 8) v -= 1.f;
    }
    Ap[i] = (__bf16)v;
}

// ---- k_wconv: lbs_weights -> bf16 (VP x 64) zero-padded ----
__global__ void k_wconv(const float* __restrict__ W, __bf16* __restrict__ Wb) {
    int i = blockIdx.x * 256 + threadIdx.x;
    if (i >= VP * 64) return;
    int jj = i % 64, v = i / 64;
    Wb[i] = (__bf16)((jj < JNUM && v < VNUM) ? W[(size_t)v * JNUM + jj] : 0.f);
}

// ---- k_posegemm: v_posed = base + exprdirs.expr + posefeat @ posedirs ------
// Block = 8 waves. Block tile: 64 rows (4 m-tiles/wave, same for all waves) x
// 128 cols (1 n-tile per wave). B panel (128 x 64K bf16, 16KB) staged into LDS
// by the TDM, double-buffered over 8 K-chunks; A-frags stream from L2.
__global__ void k_posegemm(const __bf16* __restrict__ Ap, const __bf16* __restrict__ Bt,
                           const float* __restrict__ base, const float* __restrict__ ed,
                           const float* __restrict__ expr, float* __restrict__ v_posed) {
    __shared__ __align__(16) __bf16 panel[2][128 * 64];
    int w = threadIdx.x >> 5, lane = threadIdx.x & 31;
    int ng  = blockIdx.x / MGROUPS;          // 0..245 : group of 8 n-tiles
    int mtg = blockIdx.x % MGROUPS;          // 0..7   : group of 4 m-tiles
    int h = lane >> 4, lm = lane & 15;

    unsigned long long gbase = (unsigned long long)(size_t)(Bt + (size_t)(ng * 128) * KP);
    unsigned lds0 = (unsigned)(size_t)&panel[0][0];
    unsigned lds1 = (unsigned)(size_t)&panel[1][0];

    const __bf16* aprow = Ap + (size_t)(mtg * 64 + lm) * KP;   // + mi*16*KP
    const __bf16* lrow0 = &panel[0][(w * 16 + lm) * 64];
    const __bf16* lrow1 = &panel[1][(w * 16 + lm) * 64];

    v8f acc[4] = {};
    if (w == 0) tdm_load_tile(gbase, lds0);            // chunk 0 -> buf 0

#pragma unroll
    for (int c = 0; c < 8; ++c) {
        if (w == 0) {
            if (c < 7) {
                tdm_load_tile(gbase + (unsigned long long)(c + 1) * 64ull * 2ull,
                              ((c + 1) & 1) ? lds1 : lds0);
                __builtin_amdgcn_s_wait_tensorcnt(1);  // chunk c landed
            } else {
                __builtin_amdgcn_s_wait_tensorcnt(0);
            }
        }
        __syncthreads();                               // panel[c&1] ready
        const __bf16* lrow = (c & 1) ? lrow1 : lrow0;
#pragma unroll
        for (int kk = 0; kk < 64; kk += 32) {
            bf16x16 bf = load_frag(lrow + kk + 8*h, lrow + kk + 16 + 8*h);
#pragma unroll
            for (int mi = 0; mi < 4; ++mi) {
                const __bf16* ap = aprow + (size_t)mi * 16 * KP + c * 64 + kk;
                bf16x16 af = load_frag(ap + 8*h, ap + 16 + 8*h);
                acc[mi] = __builtin_amdgcn_wmma_f32_16x16x32_bf16(
                              false, af, false, bf, (short)0, acc[mi], false, false);
            }
        }
        __syncthreads();                               // panel[c&1] consumed
    }

    int col = ng * 128 + w * 16 + lm;
    if (col < N3) {
        float basev = base[col];
        const float* edp = ed + (size_t)col * 10;
        float e0[10];
#pragma unroll
        for (int e = 0; e < 10; ++e) e0[e] = edp[e];
#pragma unroll
        for (int mi = 0; mi < 4; ++mi) {
#pragma unroll
            for (int g = 0; g < 8; ++g) {
                int b = (mtg * 4 + mi) * 16 + g + 8*h;     // C row m = g + 8*h
                const float* ex = expr + b * 10;
                float val = acc[mi][g] + basev;
#pragma unroll
                for (int e = 0; e < 10; ++e) val += e0[e] * ex[e];
                v_posed[(size_t)b * N3 + col] = val;
            }
        }
    }
}

// ---- k_skin: T = W @ A per (b, vert-tile) [WMMA], apply 3x4 transform ----
__global__ void k_skin(const __bf16* __restrict__ Wb, const __bf16* __restrict__ At,
                       const float* __restrict__ v_posed, const float* __restrict__ yoffp,
                       const float* __restrict__ gtr, float* __restrict__ out) {
    __shared__ float tileT[8][16][17];
    int w = threadIdx.x >> 5, lane = threadIdx.x & 31;
    int t  = blockIdx.x * 8 + w;
    int b  = t / VTILES, vt = t % VTILES;
    int h = lane >> 4, lm = lane & 15;
    const __bf16* wp = Wb + (size_t)(vt * 16 + lm) * 64;   // A row = vertex
    const __bf16* ap = At + (size_t)b * 1024 + lm * 64;    // B col = component
    v8f acc = {};
#pragma unroll
    for (int kb = 0; kb < 64; kb += 32) {
        bf16x16 af = load_frag(wp + kb + 8*h,  wp + kb + 16 + 8*h);
        bf16x16 bf = load_frag(ap + kb + 8*h,  ap + kb + 16 + 8*h);
        acc = __builtin_amdgcn_wmma_f32_16x16x32_bf16(
                  false, af, false, bf, (short)0, acc, false, false);
    }
#pragma unroll
    for (int g = 0; g < 8; ++g) tileT[w][g + 8*h][lm] = acc[g];
    __syncthreads();
    if (lane < 16) {
        int v = vt * 16 + lane;
        if (v < VNUM) {
            const float* vp = v_posed + (size_t)b * N3 + v * 3;
            float p0 = vp[0], p1 = vp[1], p2 = vp[2];
            float yoff = yoffp[0];
#pragma unroll
            for (int r = 0; r < 3; ++r) {
                const float* T = &tileT[w][lane][r * 4];
                float o = T[0]*p0 + T[1]*p1 + T[2]*p2 + T[3];
                if (r == 1) o += yoff;
                o += gtr[b*3 + r];
                out[(size_t)b * N3 + v*3 + r] = o;
            }
        }
    }
}

extern "C" void kernel_launch(void* const* d_in, const int* in_sizes, int n_in,
                              void* d_out, int out_size, void* d_ws, size_t ws_size,
                              hipStream_t stream) {
    const float* shape    = (const float*)d_in[0];
    const float* body     = (const float*)d_in[1];
    const float* hand     = (const float*)d_in[2];
    const float* head     = (const float*)d_in[3];
    const float* expr     = (const float*)d_in[4];
    const float* pelvis   = (const float*)d_in[5];
    const float* gtrans   = (const float*)d_in[6];
    const float* vtempl   = (const float*)d_in[7];
    const float* shapedirs= (const float*)d_in[8];
    const float* exprdirs = (const float*)d_in[9];
    const float* posedirs = (const float*)d_in[10];
    const float* lbsw     = (const float*)d_in[11];
    const float* Jreg     = (const float*)d_in[12];
    const float* hmean    = (const float*)d_in[13];
    float* out = (float*)d_out;

    // workspace carve-up (~101 MB total)
    char* ws = (char*)d_ws;
    size_t off = 0;
    auto carve = [&](size_t bytes) { size_t r = off; off = (off + bytes + 255) & ~(size_t)255; return r; };
    float*  base_shaped = (float*)(ws + carve((size_t)N3 * 4));
    float*  yoff        = (float*)(ws + carve(4));
    float*  jbase       = (float*)(ws + carve((size_t)JNUM * 3 * 4));
    float*  JE          = (float*)(ws + carve((size_t)JNUM * 30 * 4));
    float*  rotR        = (float*)(ws + carve((size_t)BNUM * JNUM * 9 * 4));
    float*  joints      = (float*)(ws + carve((size_t)BNUM * JNUM * 3 * 4));
    __bf16* At          = (__bf16*)(ws + carve((size_t)BNUM * 1024 * 2));
    __bf16* Wb          = (__bf16*)(ws + carve((size_t)VP * 64 * 2));
    __bf16* Apose       = (__bf16*)(ws + carve((size_t)BNUM * KP * 2));
    __bf16* Pdt         = (__bf16*)(ws + carve((size_t)NP2 * KP * 2));
    float*  v_posed     = (float*)(ws + carve((size_t)BNUM * N3 * 4));
    (void)in_sizes; (void)n_in; (void)out_size; (void)ws_size;

    k_base <<<(N3 + 255) / 256, 256, 0, stream>>>(vtempl, shapedirs, shape, base_shaped);
    k_yoff <<<1, 256, 0, stream>>>(vtempl, yoff);
    k_jreg <<<JNUM * 33, 256, 0, stream>>>(Jreg, base_shaped, exprdirs, jbase, JE);
    k_pose <<<(BNUM * JNUM + 255) / 256, 256, 0, stream>>>(pelvis, body, head, hand, hmean,
                                                           expr, jbase, JE, rotR, joints);
    k_fk   <<<(BNUM + 255) / 256, 256, 0, stream>>>(rotR, joints, At);
    k_pdT  <<<dim3(NP2 / 32, KP / 32), dim3(32, 8), 0, stream>>>(posedirs, Pdt);
    k_apose<<<(BNUM * KP + 255) / 256, 256, 0, stream>>>(rotR, Apose);
    k_wconv<<<(VP * 64 + 255) / 256, 256, 0, stream>>>(lbsw, Wb);
    // 246 n-groups x 8 m-groups, 8 waves/block, TDM-staged B panel
    k_posegemm<<<NGROUPS * MGROUPS, 256, 0, stream>>>(Apose, Pdt, base_shaped, exprdirs,
                                                      expr, v_posed);
    // 512 * 655 tiles, 8 waves/block
    k_skin <<<(BNUM * VTILES) / 8, 256, 0, stream>>>(Wb, At, v_posed, yoff, gtrans, out);
}
